// RelativeMoeTransformerEncoderLayer_62869731279199
// MI455X (gfx1250) — matrix operand order repaired and checked by reading, not compiled, not run on hardware
//
#include <hip/hip_runtime.h>
#include <hip/hip_bf16.h>

typedef _Float16 f16;
typedef __attribute__((ext_vector_type(16))) _Float16 v16h;
typedef __attribute__((ext_vector_type(8)))  float    v8f;

#define DEV __device__ __forceinline__

// ---------------------------------------------------------------------------
// WMMA helpers (CDNA5 16x16x32 f16 -> f32)
// ---------------------------------------------------------------------------
DEV v8f wmma32(v16h a, v16h b, v8f c) {
  return __builtin_amdgcn_wmma_f32_16x16x32_f16(false, a, false, b, (short)0, c,
                                                false, false);
}

// A-fragment (16x32, 16-bit) gather from an LDS tile stored row-major
// tile[m][kk], kk = 0..31, strideHalfs must be even.
// ISA layout: lanes 0-15: M=lane, VGPR0..3 -> K=0..7, VGPR4..7 -> K=16..23
//             lanes 16-31: M=lane-16, VGPR0..3 -> K=8..15, VGPR4..7 -> K=24..31
DEV v16h load_afrag(const f16* base, int strideHalfs, int lane) {
  int m = lane & 15, half = (lane >> 4) & 1;
  const unsigned* p = (const unsigned*)(base + (size_t)m * strideHalfs);
  union { unsigned u[8]; v16h h; } r;
  int b0 = half ? 4 : 0;    // u32 index of K pair base (K/2)
  int b1 = half ? 12 : 8;
#pragma unroll
  for (int j = 0; j < 4; ++j) { r.u[j] = p[b0 + j]; r.u[4 + j] = p[b1 + j]; }
  return r.h;
}

// B-fragment (32x16, 16-bit) gather from LDS tile stored TRANSPOSED:
// bt[n][kk], n = output column 0..15, kk = K index 0..31, strideHalfs even.
// Layout: lanes 0-15 hold K=0..7,16..23; lanes 16-31 hold K=8..15,24..31.
DEV v16h load_bfrag(const f16* base, int strideHalfs, int lane) {
  int n = lane & 15, half = (lane >> 4) & 1;
  const unsigned* p = (const unsigned*)(base + (size_t)n * strideHalfs);
  union { unsigned u[8]; v16h h; } r;
  int b0 = half ? 4 : 0;
  int b1 = half ? 12 : 8;
#pragma unroll
  for (int j = 0; j < 4; ++j) { r.u[j] = p[b0 + j]; r.u[4 + j] = p[b1 + j]; }
  return r.h;
}

// ---------------------------------------------------------------------------
// fp32 -> fp16 conversion
// ---------------------------------------------------------------------------
__global__ __launch_bounds__(256) void f32_to_f16_kernel(const float* __restrict__ in,
                                                         f16* __restrict__ out, int n) {
  int base = blockIdx.x * 1024;
#pragma unroll
  for (int i = 0; i < 4; ++i) {
    int idx = base + i * 256 + threadIdx.x;
    if (idx < n) out[idx] = (f16)in[idx];
  }
}

// ---------------------------------------------------------------------------
// LayerNorm (+ optional residual add). One block per token row of D=1024.
// ---------------------------------------------------------------------------
__global__ __launch_bounds__(256) void ln_kernel(const float* __restrict__ src,
                                                 const float* __restrict__ addin,
                                                 const float* __restrict__ gw,
                                                 const float* __restrict__ bw,
                                                 float* __restrict__ sum_out,
                                                 f16* __restrict__ y_out) {
  const int row = blockIdx.x, t = threadIdx.x;
  const size_t base = (size_t)row * 1024;
  float v[4];
  float s = 0.f;
#pragma unroll
  for (int i = 0; i < 4; ++i) {
    int c = t + i * 256;
    float x = src[base + c];
    if (addin) x += addin[base + c];
    v[i] = x; s += x;
  }
  __shared__ float red[8];
#pragma unroll
  for (int o = 1; o < 32; o <<= 1) s += __shfl_xor(s, o, 32);
  if ((t & 31) == 0) red[t >> 5] = s;
  __syncthreads();
  float tot = 0.f;
#pragma unroll
  for (int i = 0; i < 8; ++i) tot += red[i];
  float mean = tot * (1.f / 1024.f);
  __syncthreads();
  float s2 = 0.f;
#pragma unroll
  for (int i = 0; i < 4; ++i) { float d = v[i] - mean; s2 += d * d; }
#pragma unroll
  for (int o = 1; o < 32; o <<= 1) s2 += __shfl_xor(s2, o, 32);
  if ((t & 31) == 0) red[t >> 5] = s2;
  __syncthreads();
  float tot2 = 0.f;
#pragma unroll
  for (int i = 0; i < 8; ++i) tot2 += red[i];
  float rstd = rsqrtf(tot2 * (1.f / 1024.f) + 1e-5f);
#pragma unroll
  for (int i = 0; i < 4; ++i) {
    int c = t + i * 256;
    float y = (v[i] - mean) * rstd * gw[c] + bw[c];
    y_out[base + c] = (f16)y;
    if (sum_out) sum_out[base + c] = v[i];
  }
}

// ---------------------------------------------------------------------------
// Relative position encoding table: rows 0..2046 valid, row 2047 zero pad.
// pe[row][j] = sin(pos*inv_f(j)) j<512 ; cos(pos*inv_f(j-512)) j>=512
// pos = 1023 - row.
// ---------------------------------------------------------------------------
__global__ __launch_bounds__(256) void pe_kernel(f16* __restrict__ pe) {
  const int row = blockIdx.x, t = threadIdx.x;
  const size_t base = (size_t)row * 1024;
  if (row == 2047) {
#pragma unroll
    for (int i = 0; i < 4; ++i) pe[base + t + i * 256] = (f16)0.f;
    return;
  }
  const float pos = (float)(1023 - row);
#pragma unroll
  for (int i = 0; i < 4; ++i) {
    int c = t + i * 256;
    int j = (c < 512) ? c : c - 512;
    float inv = __powf(10000.f, -((float)(2 * j)) * (1.f / 1024.f));
    float ang = pos * inv;
    pe[base + c] = (f16)((c < 512) ? __sinf(ang) : __cosf(ang));
  }
}

// ---------------------------------------------------------------------------
// Generic WMMA GEMM: C[M,N] = A[M,K] @ B[K,N], A/B f16 row-major.
// Block = 256 threads (8 waves), tile 128(M) x 32(N), K step 32.
// HALF_OUT selects f16 vs f32 output.
// ---------------------------------------------------------------------------
template <bool HALF_OUT>
__global__ __launch_bounds__(256) void gemm_f16_kernel(const f16* __restrict__ A,
                                                       const f16* __restrict__ Bm,
                                                       float* __restrict__ Cf,
                                                       f16* __restrict__ Ch,
                                                       int M, int N, int K) {
  __shared__ f16 sA[128][34];
  __shared__ f16 sBt[32][34];
  const int row0 = blockIdx.y * 128;
  const int col0 = blockIdx.x * 32;
  const int t = threadIdx.x;
  const int wave = t >> 5, lane = t & 31;
  const int m16 = lane & 15, half = (lane >> 4) & 1;
  v8f acc0 = {0.f, 0.f, 0.f, 0.f, 0.f, 0.f, 0.f, 0.f};
  v8f acc1 = acc0;

  for (int k0 = 0; k0 < K; k0 += 32) {
    // Stage A tile: 128 rows x 32 halfs = 2048 u32 granules, 8/thread.
#pragma unroll
    for (int i = 0; i < 8; ++i) {
      int g = t + i * 256;
      int r = g >> 4, p = g & 15;
      unsigned v = *(const unsigned*)(A + (size_t)(row0 + r) * K + k0 + 2 * p);
      *(unsigned*)(&sA[r][2 * p]) = v;
    }
    // Stage B tile transposed: 512 u32 granules, 2/thread.
#pragma unroll
    for (int i = 0; i < 2; ++i) {
      int e = t + i * 256;
      int n2 = e & 15, kk = e >> 4;
      unsigned v = *(const unsigned*)(Bm + (size_t)(k0 + kk) * N + col0 + 2 * n2);
      union { unsigned u; f16 h[2]; } c; c.u = v;
      sBt[2 * n2][kk]     = c.h[0];
      sBt[2 * n2 + 1][kk] = c.h[1];
    }
    __syncthreads();
    v16h a  = load_afrag(&sA[wave * 16][0], 34, lane);
    v16h b0 = load_bfrag(&sBt[0][0],  34, lane);
    v16h b1 = load_bfrag(&sBt[16][0], 34, lane);
    acc0 = wmma32(a, b0, acc0);
    acc1 = wmma32(a, b1, acc1);
    __syncthreads();
  }

#pragma unroll
  for (int g = 0; g < 8; ++g) {
    int crow = row0 + wave * 16 + g + half * 8;
    size_t o0 = (size_t)crow * N + col0 + m16;
    if (HALF_OUT) {
      Ch[o0]      = (f16)acc0[g];
      Ch[o0 + 16] = (f16)acc1[g];
    } else {
      Cf[o0]      = acc0[g];
      Cf[o0 + 16] = acc1[g];
    }
  }
}

// ---------------------------------------------------------------------------
// Flash attention with relative-position bd term, fully on WMMA.
// grid = (S/64, H, B), block = 128 (4 waves). Wave w owns 16 query rows.
// q,k,v,p are f16 [2048][1024] row-major, head h at columns h*64..h*64+63.
//
// bd trick: for query tile rows i0..i0+15 and key step js..js+31 every
// needed p row is inside the 48-row window starting at i0-js+992. With
// G = q_tile @ p_window^T (16x48, 6 WMMAs), bd[m][n] = G[m][w] with
// w = m - n - 16*tt + 31, which is an intra-half lane permute of the
// WMMA C layout (rows live in VGPR index + lane half, columns in lanes).
// ---------------------------------------------------------------------------
__global__ __launch_bounds__(128) void attn_kernel(const f16* __restrict__ qg,
                                                   const f16* __restrict__ kg,
                                                   const f16* __restrict__ vg,
                                                   const f16* __restrict__ pg,
                                                   f16* __restrict__ og) {
  constexpr int S = 1024, D = 1024;
  const int b = blockIdx.z, h = blockIdx.y;
  const int t = threadIdx.x, wave = t >> 5, lane = t & 31;
  const int m16 = lane & 15, half = (lane >> 4) & 1;
  const int i0blk = blockIdx.x * 64;
  const int i0 = i0blk + wave * 16;
  const size_t rowbase = (size_t)b * S;
  const int hc = h * 64;

  __shared__ f16 kt[32][66];      // K step tile [j][hd]
  __shared__ f16 vt[64][34];      // V step tile transposed [hd][j]
  __shared__ f16 qs[4][16][66];   // per-wave Q tile [m][hd]
  __shared__ f16 pp[4][16][34];   // per-wave P probs [m][j]
  __shared__ f16 pt[96][66];      // block-wide p window [w][hd]

  // Stage this wave's Q tile (16 rows x 64 halfs = 512 u32, 16/lane).
#pragma unroll
  for (int i = 0; i < 16; ++i) {
    int g = lane + i * 32;
    int r = g >> 5, c = g & 31;
    unsigned v = *(const unsigned*)(qg + (rowbase + i0 + r) * D + hc + 2 * c);
    *(unsigned*)(&qs[wave][r][2 * c]) = v;
  }
  asm volatile("s_wait_dscnt 0" ::: "memory");
  v16h aq0 = load_afrag(&qs[wave][0][0],  66, lane);  // K dims 0..31
  v16h aq1 = load_afrag(&qs[wave][0][32], 66, lane);  // K dims 32..63

  v8f zero = {0.f, 0.f, 0.f, 0.f, 0.f, 0.f, 0.f, 0.f};
  v8f accO[4] = {zero, zero, zero, zero};
  float rowM[8], rowL[8];
#pragma unroll
  for (int g = 0; g < 8; ++g) { rowM[g] = -1e30f; rowL[g] = 0.f; }
  const float scale = 0.125f;  // 1/sqrt(64)

  for (int js = 0; js < S; js += 32) {
    __syncthreads();
    // Stage K tile: 32 rows x 32 u32, 8/thread.
#pragma unroll
    for (int i = 0; i < 8; ++i) {
      int g = t + i * 128;
      int r = g >> 5, c = g & 31;
      unsigned v = *(const unsigned*)(kg + (rowbase + js + r) * D + hc + 2 * c);
      *(unsigned*)(&kt[r][2 * c]) = v;
    }
    // Stage V tile transposed.
#pragma unroll
    for (int i = 0; i < 8; ++i) {
      int g = t + i * 128;
      int r = g >> 5, c = g & 31;
      unsigned v = *(const unsigned*)(vg + (rowbase + js + r) * D + hc + 2 * c);
      union { unsigned u; f16 h2[2]; } u2; u2.u = v;
      vt[2 * c][r]     = u2.h2[0];
      vt[2 * c + 1][r] = u2.h2[1];
    }
    // Stage p window: 96 rows x 32 u32, 24/thread. Rows always in [0,2047].
    const int pbase = i0blk - js + 992;
#pragma unroll
    for (int i = 0; i < 24; ++i) {
      int g = t + i * 128;
      int r = g >> 5, c = g & 31;
      unsigned v = *(const unsigned*)(pg + (size_t)(pbase + r) * D + hc + 2 * c);
      *(unsigned*)(&pt[r][2 * c]) = v;
    }
    __syncthreads();

    // Scores ac: two 16x16 tiles covering 32 key columns.
    v8f sc0, sc1;
    {
      v16h b0 = load_bfrag(&kt[0][0],   66, lane);
      v16h b1 = load_bfrag(&kt[0][32],  66, lane);
      v8f s = zero; s = wmma32(aq0, b0, s); s = wmma32(aq1, b1, s);
      sc0 = s;
      v16h b2 = load_bfrag(&kt[16][0],  66, lane);
      v16h b3 = load_bfrag(&kt[16][32], 66, lane);
      s = zero; s = wmma32(aq0, b2, s); s = wmma32(aq1, b3, s);
      sc1 = s;
    }

    // G = q_tile @ p_window^T : three 16-col tiles (wave window at +wave*16).
    v8f G0 = zero, G1 = zero, G2 = zero;
    {
      const f16* pw = &pt[wave * 16][0];
      v16h c0 = load_bfrag(pw,           66, lane);
      v16h c1 = load_bfrag(pw + 32,      66, lane);
      G0 = wmma32(aq0, c0, G0); G0 = wmma32(aq1, c1, G0);
      v16h c2 = load_bfrag(pw + 16 * 66, 66, lane);
      v16h c3 = load_bfrag(pw + 16 * 66 + 32, 66, lane);
      G1 = wmma32(aq0, c2, G1); G1 = wmma32(aq1, c3, G1);
      v16h c4 = load_bfrag(pw + 32 * 66, 66, lane);
      v16h c5 = load_bfrag(pw + 32 * 66 + 32, 66, lane);
      G2 = wmma32(aq0, c4, G2); G2 = wmma32(aq1, c5, G2);
    }

    // bd gather: intra-half lane permute of G, then scale.
#pragma unroll
    for (int g = 0; g < 8; ++g) {
      int m = g + half * 8;
      {
        // tile tt=0: w = m - n + 31 in [16,46] -> G1/G2 only
        int w = m - m16 + 31;
        int srcl = (w & 15) + (half << 4);
        float a1 = __shfl(G1[g], srcl, 32);
        float a2 = __shfl(G2[g], srcl, 32);
        float bdv = (w < 32) ? a1 : a2;
        sc0[g] = (sc0[g] + bdv) * scale;
      }
      {
        // tile tt=1: w = m - n + 15 in [0,30] -> G0/G1 only
        int w = m - m16 + 15;
        int srcl = (w & 15) + (half << 4);
        float a0 = __shfl(G0[g], srcl, 32);
        float a1 = __shfl(G1[g], srcl, 32);
        float bdv = (w < 16) ? a0 : a1;
        sc1[g] = (sc1[g] + bdv) * scale;
      }
    }

    // Online softmax: rows split across lane halves (C-layout), reduce over
    // the 16 lanes of each half (columns).
#pragma unroll
    for (int g = 0; g < 8; ++g) {
      float vmax = fmaxf(sc0[g], sc1[g]);
#pragma unroll
      for (int o = 1; o < 16; o <<= 1) vmax = fmaxf(vmax, __shfl_xor(vmax, o, 32));
      float nm = fmaxf(rowM[g], vmax);
      float corr = __expf(rowM[g] - nm);
      rowM[g] = nm;
      float p0 = __expf(sc0[g] - nm);
      float p1 = __expf(sc1[g] - nm);
      sc0[g] = p0; sc1[g] = p1;
      float rs = p0 + p1;
#pragma unroll
      for (int o = 1; o < 16; o <<= 1) rs += __shfl_xor(rs, o, 32);
      rowL[g] = rowL[g] * corr + rs;
      accO[0][g] *= corr; accO[1][g] *= corr;
      accO[2][g] *= corr; accO[3][g] *= corr;
    }

    // Re-layout P (C layout -> A layout) through wave-private LDS.
#pragma unroll
    for (int g = 0; g < 8; ++g) {
      int m = g + half * 8;
      pp[wave][m][m16]      = (f16)sc0[g];
      pp[wave][m][16 + m16] = (f16)sc1[g];
    }
    asm volatile("s_wait_dscnt 0" ::: "memory");
    v16h pa = load_afrag(&pp[wave][0][0], 34, lane);

    // accO += P(16x32) @ V(32x64), four 16-column tiles.
#pragma unroll
    for (int nt = 0; nt < 4; ++nt) {
      v16h bv = load_bfrag(&vt[nt * 16][0], 34, lane);
      accO[nt] = wmma32(pa, bv, accO[nt]);
    }
  }

  // Epilogue: normalize and store f16.
#pragma unroll
  for (int nt = 0; nt < 4; ++nt) {
#pragma unroll
    for (int g = 0; g < 8; ++g) {
      int row = i0 + g + half * 8;
      int col = hc + nt * 16 + m16;
      og[(rowbase + row) * D + col] = (f16)(accO[nt][g] / rowL[g]);
    }
  }
}

// ---------------------------------------------------------------------------
// Router: sel = sigmoid(x2 @ sel_w), top-4. One wave per token; lane = expert.
// ---------------------------------------------------------------------------
__global__ __launch_bounds__(128) void sel_topk_kernel(const f16* __restrict__ x2,
                                                       const float* __restrict__ selw,
                                                       float* __restrict__ gates,
                                                       int* __restrict__ eidx) {
  const int token = blockIdx.x * 4 + (threadIdx.x >> 5);
  const int lane = threadIdx.x & 31;
  const f16* xr = x2 + (size_t)token * 1024;
  float dot = 0.f;
  for (int d = 0; d < 1024; ++d)
    dot += (float)xr[d] * selw[(size_t)d * 32 + lane];
  float s = 1.f / (1.f + __expf(-dot));
#pragma unroll
  for (int k = 0; k < 4; ++k) {
    float m = s;
#pragma unroll
    for (int o = 1; o < 32; o <<= 1) m = fmaxf(m, __shfl_xor(m, o, 32));
    unsigned long long msk = __ballot(s == m);
    int src = __ffsll((long long)msk) - 1;
    if (lane == 0) { gates[token * 4 + k] = m; eidx[token * 4 + k] = src; }
    if (lane == src) s = -1e30f;
  }
}

// ---------------------------------------------------------------------------
// MoE (top-4 experts only) + final residual. One block (256) per token.
// keys fp32 [E][D][ES], values fp32 [E][ES][D].
// ---------------------------------------------------------------------------
__global__ __launch_bounds__(256) void moe_kernel(const f16* __restrict__ x2,
                                                  const float* __restrict__ src2,
                                                  const float* __restrict__ gates,
                                                  const int* __restrict__ eidx,
                                                  const float* __restrict__ keys,
                                                  const float* __restrict__ vals,
                                                  float* __restrict__ out) {
  const int token = blockIdx.x, t = threadIdx.x;
  __shared__ float xs[1024];
  __shared__ float hpart[2][128];
  __shared__ float hbuf[128];
  const size_t base = (size_t)token * 1024;
  float acc[4];
#pragma unroll
  for (int i = 0; i < 4; ++i) {
    int c = t + i * 256;
    xs[c] = (float)x2[base + c];
    acc[i] = src2[base + c];
  }
  __syncthreads();
  for (int k = 0; k < 4; ++k) {
    int e = eidx[token * 4 + k];
    float g = gates[token * 4 + k];
    int j = t & 127, hseg = t >> 7;
    const float* kcol = keys + ((size_t)e * 1024 + hseg * 512) * 128 + j;
    float hp = 0.f;
    for (int d = 0; d < 512; ++d) hp += xs[hseg * 512 + d] * kcol[(size_t)d * 128];
    hpart[hseg][j] = hp;
    __syncthreads();
    if (t < 128) hbuf[t] = fmaxf(hpart[0][t] + hpart[1][t], 0.f) * g;
    __syncthreads();
    const float* vrow = vals + (size_t)e * 128 * 1024;
#pragma unroll
    for (int i = 0; i < 4; ++i) {
      int d = t + i * 256;
      float a = 0.f;
      for (int jj = 0; jj < 128; ++jj) a += hbuf[jj] * vrow[(size_t)jj * 1024 + d];
      acc[i] += a;
    }
    __syncthreads();
  }
#pragma unroll
  for (int i = 0; i < 4; ++i) out[base + t + i * 256] = acc[i];
}

// ---------------------------------------------------------------------------
// Launch
// ---------------------------------------------------------------------------
extern "C" void kernel_launch(void* const* d_in, const int* in_sizes, int n_in,
                              void* d_out, int out_size, void* d_ws, size_t ws_size,
                              hipStream_t stream) {
  (void)in_sizes; (void)n_in; (void)out_size; (void)ws_size;
  const float* src  = (const float*)d_in[0];
  const float* wq   = (const float*)d_in[1];
  const float* wk   = (const float*)d_in[2];
  const float* wv   = (const float*)d_in[3];
  const float* wp   = (const float*)d_in[4];
  const float* wo   = (const float*)d_in[5];
  const float* ln1g = (const float*)d_in[6];
  const float* ln1b = (const float*)d_in[7];
  const float* ln2g = (const float*)d_in[8];
  const float* ln2b = (const float*)d_in[9];
  const float* selw = (const float*)d_in[10];
  const float* keys = (const float*)d_in[11];
  const float* vals = (const float*)d_in[12];
  float* out = (float*)d_out;
  char* ws = (char*)d_ws;
  const size_t MB = 1024 * 1024;

  f16* xh   = (f16*)(ws + 0);        // 4MB; later reused as attn output (f16)
  f16* wqh  = (f16*)(ws + 4 * MB);   // 2MB
  f16* wkh  = (f16*)(ws + 6 * MB);
  f16* wvh  = (f16*)(ws + 8 * MB);
  f16* wph  = (f16*)(ws + 10 * MB);
  f16* woh  = (f16*)(ws + 12 * MB);
  f16* qh   = (f16*)(ws + 14 * MB);  // 4MB
  f16* kh   = (f16*)(ws + 18 * MB);
  f16* vh   = (f16*)(ws + 22 * MB);
  f16* peh  = (f16*)(ws + 26 * MB);
  f16* ph   = (f16*)(ws + 30 * MB);  // total 34MB
  // Aliases (lifetimes are disjoint, kernels run sequentially on stream):
  f16*   attnh    = xh;                        // after QKV/p GEMMs done
  float* attnproj = (float*)(ws + 4 * MB);     // over wqh..wph (8MB)
  float* src2     = (float*)(ws + 14 * MB);    // over qh..kh (8MB)
  f16*   x2h      = (f16*)(ws + 22 * MB);      // over vh
  float* gates    = (float*)(ws + 30 * MB);    // over ph
  int*   eidx     = (int*)(ws + 30 * MB + 64 * 1024);

  const int DD = 1024 * 1024;
  f32_to_f16_kernel<<<1024, 256, 0, stream>>>(wq, wqh, DD);
  f32_to_f16_kernel<<<1024, 256, 0, stream>>>(wk, wkh, DD);
  f32_to_f16_kernel<<<1024, 256, 0, stream>>>(wv, wvh, DD);
  f32_to_f16_kernel<<<1024, 256, 0, stream>>>(wp, wph, DD);
  f32_to_f16_kernel<<<1024, 256, 0, stream>>>(wo, woh, DD);

  ln_kernel<<<2048, 256, 0, stream>>>(src, nullptr, ln1g, ln1b, nullptr, xh);
  pe_kernel<<<2048, 256, 0, stream>>>(peh);

  dim3 gg(1024 / 32, 2048 / 128);
  gemm_f16_kernel<true><<<gg, 256, 0, stream>>>(xh, wqh, nullptr, qh, 2048, 1024, 1024);
  gemm_f16_kernel<true><<<gg, 256, 0, stream>>>(xh, wkh, nullptr, kh, 2048, 1024, 1024);
  gemm_f16_kernel<true><<<gg, 256, 0, stream>>>(xh, wvh, nullptr, vh, 2048, 1024, 1024);
  gemm_f16_kernel<true><<<gg, 256, 0, stream>>>(peh, wph, nullptr, ph, 2048, 1024, 1024);

  attn_kernel<<<dim3(16, 16, 2), 128, 0, stream>>>(qh, kh, vh, ph, attnh);

  gemm_f16_kernel<false><<<gg, 256, 0, stream>>>(attnh, woh, attnproj, nullptr,
                                                 2048, 1024, 1024);

  ln_kernel<<<2048, 256, 0, stream>>>(src, attnproj, ln2g, ln2b, src2, x2h);
  sel_topk_kernel<<<512, 128, 0, stream>>>(x2h, selw, gates, eidx);
  moe_kernel<<<2048, 256, 0, stream>>>(x2h, src2, gates, eidx, keys, vals, out);
}
